// MultiHeadLatentAttention_49039936586411
// MI455X (gfx1250) — matrix operand order repaired and checked by reading, not compiled, not run on hardware
//
#include <hip/hip_runtime.h>

// ---------------- problem constants ----------------
constexpr int Bb = 4;
constexpr int Cc = 256;
constexpr int Hh = 48, Ww = 48;
constexpr int Nn = Hh * Ww;          // 2304
constexpr int NH = 8;
constexpr int HD = 32;               // head dim
constexpr int LD = 32;               // latent dim
constexpr float SCALE = 0.17677669529663687f; // 32^-0.5

typedef __attribute__((ext_vector_type(16))) __bf16 v16bf;
typedef __attribute__((ext_vector_type(8)))  float  v8f;

// ---------------- bf16 helpers ----------------
__device__ __forceinline__ unsigned short f2bf(float f) {
    union { float f; unsigned u; } v; v.f = f;
    unsigned u = v.u;
    unsigned r = u + 0x7fffu + ((u >> 16) & 1u);   // RNE
    return (unsigned short)(r >> 16);
}

union TileU { uint4 q[2]; v16bf v; };

// A-operand: 16x32 bf16, per ISA: lanes 0-15 hold K = {0..7, 16..23},
// lanes 16-31 hold K = {8..15, 24..31} (two split 8-elem chunks).
__device__ __forceinline__ v16bf load_a16x32(const unsigned short* base, int stride,
                                             int k0, int lane) {
    int hl = (lane >> 4) & 1;
    int r  = lane & 15;
    const unsigned short* p = base + (size_t)r * stride + k0 + hl * 8;
    TileU t;
    t.q[0] = *reinterpret_cast<const uint4*>(p);
    t.q[1] = *reinterpret_cast<const uint4*>(p + 16);
    return t.v;
}

// B-operand: 32x16 bf16, same K order as A: lanes 0-15 hold K = {0..7, 16..23},
// lanes 16-31 hold K = {8..15, 24..31} (two split 8-elem chunks).
// Lane holds column n = lane&15; caller's "base" is row-major [row][k],
// row = matrix column of the B operand.
__device__ __forceinline__ v16bf load_b32x16(const unsigned short* base, int stride,
                                             int k0, int lane) {
    int hl = (lane >> 4) & 1;
    int r  = lane & 15;
    const unsigned short* p = base + (size_t)r * stride + k0 + hl * 8;
    TileU t;
    t.q[0] = *reinterpret_cast<const uint4*>(p);
    t.q[1] = *reinterpret_cast<const uint4*>(p + 16);
    return t.v;
}

__device__ __forceinline__ v8f wmma_bf16(v16bf a, v16bf b, v8f c) {
    return __builtin_amdgcn_wmma_f32_16x16x32_bf16(false, a, false, b,
                                                   (short)0, c, false, false);
}

// ---------------- kernel 0: transpose-convert x -> xbT[b][n][c] (bf16) ----------------
__global__ void k_prep_x(const float* __restrict__ x, unsigned short* __restrict__ xbT) {
    int idx = blockIdx.x * blockDim.x + threadIdx.x;
    if (idx >= Bb * Cc * Nn) return;
    int b = idx / (Cc * Nn);
    int rem = idx - b * (Cc * Nn);
    int c = rem / Nn;
    int n = rem - c * Nn;
    xbT[((size_t)b * Nn + n) * Cc + c] = f2bf(x[idx]);
}

// ---------------- kernel 1: out_w -> bf16 ----------------
__global__ void k_prep_outw(const float* __restrict__ w, unsigned short* __restrict__ wb) {
    int idx = blockIdx.x * blockDim.x + threadIdx.x;
    if (idx < Cc * Cc) wb[idx] = f2bf(w[idx]);
}

// ---------------- kernel 2: fused weights Wall (768x256 bf16) ----------------
// rows 0..255   : Wlq[h*32+l, c] = sum_d latent_w[l,d] * q_w[h*32+d, c]
// rows 256..511 : Wlk from kv_w rows 0..255
// rows 512..767 : v weights = kv_w rows 256..511
__global__ void k_fuse_w(const float* __restrict__ q_w, const float* __restrict__ kv_w,
                         const float* __restrict__ latent_w,
                         unsigned short* __restrict__ Wall) {
    int idx = blockIdx.x * blockDim.x + threadIdx.x;
    if (idx >= 768 * Cc) return;
    int row = idx / Cc;
    int c   = idx - row * Cc;
    float val;
    if (row < 512) {
        int rr = (row < 256) ? row : (row - 256);
        const float* wsrc = (row < 256) ? q_w : kv_w;
        int h = rr >> 5, l = rr & 31;
        float acc = 0.f;
        #pragma unroll
        for (int d = 0; d < 32; ++d)
            acc += latent_w[l * HD + d] * wsrc[(size_t)(h * 32 + d) * Cc + c];
        val = acc;
    } else {
        val = kv_w[(size_t)(256 + row - 512) * Cc + c];
    }
    Wall[(size_t)row * Cc + c] = f2bf(val);
}

// ---------------- kernel 3: QKV GEMM  Y[b] = Wall(768x256) @ x[b]  ----------------
// Output scattered into lq[b][h][n][l], lk[b][h][n][l], v[b][h][d][n] (all bf16).
__global__ void k_qkv_gemm(const unsigned short* __restrict__ Wall,
                           const unsigned short* __restrict__ xbT,
                           unsigned short* __restrict__ lq,
                           unsigned short* __restrict__ lk,
                           unsigned short* __restrict__ vv) {
    int tile = blockIdx.x;
    int nt = tile % (Nn / 16);
    int mt = (tile / (Nn / 16)) % 48;
    int b  = tile / ((Nn / 16) * 48);
    int lane = threadIdx.x;

    const unsigned short* Abase = Wall + (size_t)mt * 16 * Cc;
    const unsigned short* Bbase = xbT + (size_t)b * Nn * Cc + (size_t)nt * 16 * Cc;

    v8f acc = {0.f, 0.f, 0.f, 0.f, 0.f, 0.f, 0.f, 0.f};
    #pragma unroll
    for (int kc = 0; kc < Cc / 32; ++kc) {
        v16bf a = load_a16x32(Abase, Cc, kc * 32, lane);
        v16bf bm = load_b32x16(Bbase, Cc, kc * 32, lane);
        acc = wmma_bf16(a, bm, acc);
    }

    int hl = (lane >> 4) & 1;
    int ncol = nt * 16 + (lane & 15);
    #pragma unroll
    for (int r = 0; r < 8; ++r) {
        int m = mt * 16 + r + hl * 8;
        unsigned short bv = f2bf(acc[r]);
        if (m < 256) {
            int h = m >> 5, l = m & 31;
            lq[(((size_t)b * NH + h) * Nn + ncol) * LD + l] = bv;
        } else if (m < 512) {
            int mm = m - 256;
            int h = mm >> 5, l = mm & 31;
            lk[(((size_t)b * NH + h) * Nn + ncol) * LD + l] = bv;
        } else {
            int cch = m - 512;
            int h = cch >> 5, d = cch & 31;
            vv[(((size_t)b * NH + h) * HD + d) * Nn + ncol] = bv;
        }
    }
}

// ---------------- kernel 4: flash attention (one wave = 16 queries) ----------------
// lq/lk: [b][h][n][l] bf16, v: [b][h][d][n] bf16, obf: [b][n][h*32+d] bf16
__global__ void k_attn(const unsigned short* __restrict__ lq,
                       const unsigned short* __restrict__ lk,
                       const unsigned short* __restrict__ vv,
                       unsigned short* __restrict__ obf) {
    __shared__ unsigned short Plds[16 * 32];

    int id = blockIdx.x;
    int qt = id % (Nn / 16);
    int h  = (id / (Nn / 16)) % NH;
    int b  = id / ((Nn / 16) * NH);
    int lane = threadIdx.x;
    int hl = (lane >> 4) & 1;
    int r15 = lane & 15;

    const unsigned short* lqb = lq + ((size_t)b * NH + h) * Nn * LD;
    const unsigned short* lkb = lk + ((size_t)b * NH + h) * Nn * LD;
    const unsigned short* vb  = vv + ((size_t)b * NH + h) * HD * Nn;

    // Q tile: 16 queries x 32 latent, loaded once (K = full latent dim)
    v16bf aQ = load_a16x32(lqb + (size_t)qt * 16 * LD, LD, 0, lane);

    float mrow[8], lrow[8];
    #pragma unroll
    for (int r = 0; r < 8; ++r) { mrow[r] = -1e30f; lrow[r] = 0.f; }
    v8f o0 = {0.f,0.f,0.f,0.f,0.f,0.f,0.f,0.f};
    v8f o1 = {0.f,0.f,0.f,0.f,0.f,0.f,0.f,0.f};
    const v8f vz = {0.f,0.f,0.f,0.f,0.f,0.f,0.f,0.f};

    for (int kb = 0; kb < Nn / 32; ++kb) {
        int m0 = kb * 32;
        // scores: S = lq (16x32) @ lk^T (32x16), two key half-tiles
        v16bf bk0 = load_b32x16(lkb + (size_t)m0 * LD, LD, 0, lane);
        v16bf bk1 = load_b32x16(lkb + (size_t)(m0 + 16) * LD, LD, 0, lane);
        v8f s0 = wmma_bf16(aQ, bk0, vz);
        v8f s1 = wmma_bf16(aQ, bk1, vz);

        float e0[8], e1[8];
        #pragma unroll
        for (int r = 0; r < 8; ++r) {
            float p0 = s0[r] * SCALE;
            float p1 = s1[r] * SCALE;
            // row max across the 16 lanes of this half
            float t = fmaxf(p0, p1);
            #pragma unroll
            for (int mask = 1; mask < 16; mask <<= 1)
                t = fmaxf(t, __shfl_xor(t, mask));
            float mnew  = fmaxf(mrow[r], t);
            float alpha = __expf(mrow[r] - mnew);
            float a0 = __expf(p0 - mnew);
            float a1 = __expf(p1 - mnew);
            float rs = a0 + a1;
            #pragma unroll
            for (int mask = 1; mask < 16; mask <<= 1)
                rs += __shfl_xor(rs, mask);
            lrow[r] = lrow[r] * alpha + rs;
            mrow[r] = mnew;
            o0[r] *= alpha;
            o1[r] *= alpha;
            e0[r] = a0; e1[r] = a1;
        }

        // D-layout -> A-layout transpose of P through LDS
        #pragma unroll
        for (int r = 0; r < 8; ++r) {
            int qrow = r + hl * 8;
            Plds[qrow * 32 + r15]      = f2bf(e0[r]);
            Plds[qrow * 32 + 16 + r15] = f2bf(e1[r]);
        }
        __syncthreads();
        v16bf aP = load_a16x32(Plds, 32, 0, lane);
        __syncthreads();

        // O += P (16x32) @ V'(32x16)  for each half of head-dim
        v16bf bv0 = load_b32x16(vb, Nn, m0, lane);                       // d = 0..15
        v16bf bv1 = load_b32x16(vb + (size_t)16 * Nn, Nn, m0, lane);     // d = 16..31
        o0 = wmma_bf16(aP, bv0, o0);
        o1 = wmma_bf16(aP, bv1, o1);
    }

    #pragma unroll
    for (int r = 0; r < 8; ++r) {
        float inv = 1.0f / lrow[r];
        int n = qt * 16 + r + hl * 8;
        size_t base = ((size_t)b * Nn + n) * Cc + h * 32;
        obf[base + r15]      = f2bf(o0[r] * inv);
        obf[base + 16 + r15] = f2bf(o1[r] * inv);
    }
}

// ---------------- kernel 5: out-proj  out = out_w @ O + x ----------------
__global__ void k_outproj(const unsigned short* __restrict__ ow,
                          const unsigned short* __restrict__ obf,
                          const float* __restrict__ x,
                          float* __restrict__ out) {
    int tile = blockIdx.x;
    int nt = tile % (Nn / 16);
    int ot = (tile / (Nn / 16)) % (Cc / 16);
    int b  = tile / ((Nn / 16) * (Cc / 16));
    int lane = threadIdx.x;

    const unsigned short* Abase = ow + (size_t)ot * 16 * Cc;
    const unsigned short* Bbase = obf + (size_t)b * Nn * Cc + (size_t)nt * 16 * Cc;

    v8f acc = {0.f,0.f,0.f,0.f,0.f,0.f,0.f,0.f};
    #pragma unroll
    for (int kc = 0; kc < Cc / 32; ++kc) {
        v16bf a = load_a16x32(Abase, Cc, kc * 32, lane);
        v16bf bm = load_b32x16(Bbase, Cc, kc * 32, lane);
        acc = wmma_bf16(a, bm, acc);
    }

    int hl = (lane >> 4) & 1;
    int n = nt * 16 + (lane & 15);
    #pragma unroll
    for (int r = 0; r < 8; ++r) {
        int o = ot * 16 + r + hl * 8;
        size_t idx = ((size_t)b * Cc + o) * Nn + n;
        out[idx] = acc[r] + x[idx];
    }
}

// ---------------- host ----------------
extern "C" void kernel_launch(void* const* d_in, const int* in_sizes, int n_in,
                              void* d_out, int out_size, void* d_ws, size_t ws_size,
                              hipStream_t stream) {
    const float* x        = (const float*)d_in[0];
    const float* q_w      = (const float*)d_in[1];
    const float* kv_w     = (const float*)d_in[2];
    const float* latent_w = (const float*)d_in[3];
    const float* out_w    = (const float*)d_in[4];
    float* out = (float*)d_out;

    unsigned char* ws = (unsigned char*)d_ws;
    // bf16 workspace layout
    unsigned short* Wall = (unsigned short*)(ws + 0);                 // 768*256
    unsigned short* OW   = (unsigned short*)(ws + 393216);            // 256*256
    unsigned short* XBT  = (unsigned short*)(ws + 524288);            // B*N*C
    unsigned short* LQ   = (unsigned short*)(ws + 5242880);           // B*H*N*L
    unsigned short* LK   = (unsigned short*)(ws + 9961472);
    unsigned short* VV   = (unsigned short*)(ws + 14680064);          // B*H*D*N
    unsigned short* OBF  = (unsigned short*)(ws + 19398656);          // B*N*C

    k_prep_x   <<<(Bb * Cc * Nn + 255) / 256, 256, 0, stream>>>(x, XBT);
    k_prep_outw<<<(Cc * Cc + 255) / 256,      256, 0, stream>>>(out_w, OW);
    k_fuse_w   <<<(768 * Cc + 255) / 256,     256, 0, stream>>>(q_w, kv_w, latent_w, Wall);

    k_qkv_gemm <<<Bb * 48 * (Nn / 16), 32, 0, stream>>>(Wall, XBT, LQ, LK, VV);
    k_attn     <<<Bb * NH * (Nn / 16), 32, 0, stream>>>(LQ, LK, VV, OBF);
    k_outproj  <<<Bb * (Cc / 16) * (Nn / 16), 32, 0, stream>>>(OW, OBF, x, out);
}
